// PVConv_65953517797928
// MI455X (gfx1250) — matrix operand order repaired and checked by reading, not compiled, not run on hardware
//
#include <hip/hip_runtime.h>
#include <cstdint>

typedef _Float16 half_t;
typedef __attribute__((ext_vector_type(16))) _Float16 v16h;
typedef __attribute__((ext_vector_type(8)))  _Float16 v8h;
typedef __attribute__((ext_vector_type(8)))  float    v8f;

#define BB    8
#define CIN   64
#define COUT  64
#define NPTS  32768
#define RR    32
#define R3    32768
#define SLOPE 0.1f
#define BN_EPS 1e-4f
#define EPSF  1e-6f

#define WFRAG_HALFS (27 * 2 * 4 * 512)            // 110592 halfs
#define WFRAG_BYTES (WFRAG_HALFS * 2)             // 216 KB

// ---------------------------------------------------------------------------
// Trilinear corner computation (shared by scatter & gather)
// ---------------------------------------------------------------------------
struct Corners { int idx[8]; float w[8]; };

__device__ inline void compute_corners(const float* __restrict__ coords,
                                       const float* __restrict__ stats,
                                       int b, int n, Corners& c) {
  const float* cb = coords + (size_t)b * 3 * NPTS;
  float mx = stats[b * 4 + 0], my = stats[b * 4 + 1], mz = stats[b * 4 + 2];
  float s  = stats[b * 4 + 3];
  float vx = fminf(fmaxf((cb[n]            - mx) * s + 16.0f, 0.0f), 31.0f);
  float vy = fminf(fmaxf((cb[NPTS + n]     - my) * s + 16.0f, 0.0f), 31.0f);
  float vz = fminf(fmaxf((cb[2 * NPTS + n] - mz) * s + 16.0f, 0.0f), 31.0f);
  int ix = (int)floorf(vx), iy = (int)floorf(vy), iz = (int)floorf(vz);
  float fx = vx - (float)ix, fy = vy - (float)iy, fz = vz - (float)iz;
#pragma unroll
  for (int k = 0; k < 8; ++k) {
    int dx = (k >> 2) & 1, dy = (k >> 1) & 1, dz = k & 1;
    int xi = min(ix + dx, 31), yi = min(iy + dy, 31), zi = min(iz + dz, 31);
    float w = (dx ? fx : 1.0f - fx) * (dy ? fy : 1.0f - fy) * (dz ? fz : 1.0f - fz);
    c.idx[k] = (xi * 32 + yi) * 32 + zi;
    c.w[k]   = w;
  }
}

// ---------------------------------------------------------------------------
// Per-batch stats: mean xyz + scale = R / (2*maxnorm*(1+eps))
// ---------------------------------------------------------------------------
__global__ __launch_bounds__(256) void pv_stats(const float* __restrict__ coords,
                                                float* __restrict__ stats) {
  __shared__ float red[256];
  __shared__ float mvals[3];
  int b = blockIdx.x, t = threadIdx.x;
  const float* cb = coords + (size_t)b * 3 * NPTS;
  for (int axis = 0; axis < 3; ++axis) {
    float s = 0.0f;
    for (int n = t; n < NPTS; n += 256) s += cb[axis * NPTS + n];
    red[t] = s; __syncthreads();
    for (int o = 128; o > 0; o >>= 1) { if (t < o) red[t] += red[t + o]; __syncthreads(); }
    if (t == 0) mvals[axis] = red[0] / (float)NPTS;
    __syncthreads();
  }
  float mx = mvals[0], my = mvals[1], mz = mvals[2];
  float mn = 0.0f;
  for (int n = t; n < NPTS; n += 256) {
    float dx = cb[n] - mx, dy = cb[NPTS + n] - my, dz = cb[2 * NPTS + n] - mz;
    mn = fmaxf(mn, sqrtf(dx * dx + dy * dy + dz * dz));
  }
  red[t] = mn; __syncthreads();
  for (int o = 128; o > 0; o >>= 1) { if (t < o) red[t] = fmaxf(red[t], red[t + o]); __syncthreads(); }
  if (t == 0) {
    stats[b * 4 + 0] = mx; stats[b * 4 + 1] = my; stats[b * 4 + 2] = mz;
    stats[b * 4 + 3] = (float)RR / (2.0f * red[0] * (1.0f + EPSF));
  }
}

// ---------------------------------------------------------------------------
// Transpose features [B,64,N] -> [B,N,64] via LDS tiles
// ---------------------------------------------------------------------------
__global__ __launch_bounds__(256) void pv_transpose(const float* __restrict__ feat,
                                                    float* __restrict__ featT) {
  __shared__ float tile[64][33];
  int t = threadIdx.x;
  int bn = blockIdx.x;                 // B * (N/32)
  int b  = bn / (NPTS / 32);
  int n0 = (bn % (NPTS / 32)) * 32;
#pragma unroll
  for (int k = 0; k < 8; ++k) {
    int e = k * 256 + t; int c = e >> 5, nn = e & 31;
    tile[c][nn] = feat[(size_t)b * 64 * NPTS + (size_t)c * NPTS + n0 + nn];
  }
  __syncthreads();
#pragma unroll
  for (int k = 0; k < 8; ++k) {
    int e = k * 256 + t; int nn = e >> 6, c = e & 63;
    featT[((size_t)b * NPTS + n0 + nn) * 64 + c] = tile[c][nn];
  }
}

// ---------------------------------------------------------------------------
// Fold conv-bias + BN into scale/shift for all three branches
// sc layout: [scale1|shift1|scale2|shift2|scalep|shiftp], 64 each
// ---------------------------------------------------------------------------
__global__ void pv_bnfold(const float* cb1, const float* g1, const float* b1, const float* m1, const float* v1,
                          const float* cb2, const float* g2, const float* b2, const float* m2, const float* v2,
                          const float* mb,  const float* gp, const float* bp, const float* mp, const float* vp,
                          float* sc) {
  int c = threadIdx.x;
  float s1 = g1[c] * rsqrtf(v1[c] + BN_EPS);
  sc[c] = s1;        sc[64  + c] = (cb1[c] - m1[c]) * s1 + b1[c];
  float s2 = g2[c] * rsqrtf(v2[c] + BN_EPS);
  sc[128 + c] = s2;  sc[192 + c] = (cb2[c] - m2[c]) * s2 + b2[c];
  float sp = gp[c] * rsqrtf(vp[c] + BN_EPS);
  sc[256 + c] = sp;  sc[320 + c] = (mb[c] - mp[c]) * sp + bp[c];
}

// ---------------------------------------------------------------------------
// Pre-swizzle conv weights [O=64][I=64][27] f32 into per-lane WMMA B-fragments:
// wfrag[tap][kstep][ntile][lane][16] f16, lane n=l&15 holds K = 32*s+8*(l>>4)+j (j<8), 32*s+16+8*(l>>4)+(j-8) (j>=8)
// ---------------------------------------------------------------------------
__global__ void pv_wswizzle(const float* __restrict__ w, half_t* __restrict__ wfrag) {
  int idx = blockIdx.x * 256 + threadIdx.x;
  if (idx >= WFRAG_HALFS) return;
  int j    =  idx        & 15;
  int lane = (idx >> 4)  & 31;
  int nt   = (idx >> 9)  & 3;
  int s    = (idx >> 11) & 1;
  int tap  =  idx >> 12;
  int o = nt * 16 + (lane & 15);
  int k = 32 * s + 8 * (lane >> 4) + (j < 8 ? j : 8 + j);
  wfrag[idx] = (half_t)w[((size_t)o * 64 + k) * 27 + tap];
}

// ---------------------------------------------------------------------------
// Trilinear scatter-voxelize: 64 lanes = channels, 4 points per block
// ---------------------------------------------------------------------------
__global__ __launch_bounds__(256) void pv_scatter(const float* __restrict__ featT,
                                                  const float* __restrict__ coords,
                                                  const float* __restrict__ stats,
                                                  float* __restrict__ acc,
                                                  float* __restrict__ cnt) {
  int t = threadIdx.x;
  int c = t & 63;
  int g = blockIdx.x * 4 + (t >> 6);      // global point id over B*N
  int b = g / NPTS, n = g - b * NPTS;
  float f = featT[(size_t)g * 64 + c];
  Corners cr; compute_corners(coords, stats, b, n, cr);
#pragma unroll
  for (int k = 0; k < 8; ++k) {
    size_t base = ((size_t)(b * R3 + cr.idx[k])) * 64;
    unsafeAtomicAdd(&acc[base + c], cr.w[k] * f);
    if (c == 0) unsafeAtomicAdd(&cnt[b * R3 + cr.idx[k]], cr.w[k]);
  }
}

// ---------------------------------------------------------------------------
// Normalize by weight-count and convert to f16 voxel-major [B*R3, 64]
// ---------------------------------------------------------------------------
__global__ __launch_bounds__(256) void pv_normalize(const float* __restrict__ acc,
                                                    const float* __restrict__ cnt,
                                                    half_t* __restrict__ vox) {
  size_t i = (size_t)blockIdx.x * 256 + threadIdx.x;
  size_t v = i >> 6;
  float cv = fmaxf(cnt[v], 1e-8f);
  vox[i] = (half_t)(acc[i] / cv);
}

// ---------------------------------------------------------------------------
// A-fragment loader: 16 spatial rows x 32 input channels, zero-padded OOB.
// Per-lane: halfs 0..7 -> K = 32s+8h.., halfs 8..15 -> K = 32s+16+8h..
// ---------------------------------------------------------------------------
__device__ inline v16h load_a(const half_t* __restrict__ vin, size_t bbase,
                              int xn, int yn, int zn, bool okxy, int chbase) {
  v16h a;
  bool ok = okxy & ((unsigned)zn < 32u);
  if (ok) {
    const half_t* p = vin + (bbase + (size_t)((xn * 32 + yn) * 32 + zn)) * 64 + chbase;
    v8h c0 = *(const v8h*)p;
    v8h c1 = *(const v8h*)(p + 16);
#pragma unroll
    for (int j = 0; j < 8; ++j) { a[j] = c0[j]; a[8 + j] = c1[j]; }
  } else {
#pragma unroll
    for (int j = 0; j < 16; ++j) a[j] = (_Float16)0;
  }
  return a;
}

// ---------------------------------------------------------------------------
// Implicit-GEMM 3x3x3 conv via v_wmma_f32_16x16x32_f16.
// Weights staged in LDS (216KB). Each wave owns one (b,x,y) z-column:
// 2 spatial tiles of 16 z-voxels x 64 out channels = 8 accumulators.
// Fully unrolled 27 taps x 2 K-steps; 8 WMMAs per fragment group.
// ---------------------------------------------------------------------------
template <bool OUT_F16>
__global__ __launch_bounds__(256) void pv_conv3d(const half_t* __restrict__ vin,
                                                 const half_t* __restrict__ wfrag,
                                                 const float* __restrict__ scale,
                                                 const float* __restrict__ shift,
                                                 void* __restrict__ outp) {
  extern __shared__ char smem_raw[];
  half_t* swf = (half_t*)smem_raw;

  // cooperative stage of all weight fragments into LDS (16B chunks)
  for (int e = threadIdx.x; e < WFRAG_HALFS / 8; e += 256) {
    *(v8h*)(swf + (size_t)e * 8) = *(const v8h*)(wfrag + (size_t)e * 8);
  }
  __syncthreads();

  int wave = threadIdx.x >> 5;
  int lane = threadIdx.x & 31;
  int col  = blockIdx.x * 8 + wave;            // over B*32*32 (b,x,y) columns
  int b  = col >> 10;
  int xy = col & 1023;
  int x  = xy >> 5, y = xy & 31;
  int h   = lane >> 4;
  int row = lane & 15;
  size_t bbase = (size_t)b * R3;

  v8f acc[2][4];
#pragma unroll
  for (int m = 0; m < 2; ++m)
#pragma unroll
    for (int nt = 0; nt < 4; ++nt)
#pragma unroll
      for (int r = 0; r < 8; ++r) acc[m][nt][r] = 0.0f;

#pragma unroll
  for (int t = 0; t < 27; ++t) {
    const int dz = t % 3 - 1, dy = (t / 3) % 3 - 1, dx = t / 9 - 1;
    int xn = x + dx, yn = y + dy;
    bool okxy = ((unsigned)xn < 32u) & ((unsigned)yn < 32u);
#pragma unroll
    for (int s = 0; s < 2; ++s) {
      int chbase = 32 * s + 8 * h;
      // issue all loads first: 2 A fragments (global) + 4 B fragments (LDS)
      v16h a0 = load_a(vin, bbase, xn, yn,      row + dz, okxy, chbase);
      v16h a1 = load_a(vin, bbase, xn, yn, 16 + row + dz, okxy, chbase);
      const half_t* wb = swf + ((size_t)(t * 2 + s) * 4) * 512 + lane * 16;
      v16h bf0 = *(const v16h*)(wb);
      v16h bf1 = *(const v16h*)(wb + 512);
      v16h bf2 = *(const v16h*)(wb + 1024);
      v16h bf3 = *(const v16h*)(wb + 1536);
      acc[0][0] = __builtin_amdgcn_wmma_f32_16x16x32_f16(false, a0, false, bf0, (short)0, acc[0][0], false, false);
      acc[0][1] = __builtin_amdgcn_wmma_f32_16x16x32_f16(false, a0, false, bf1, (short)0, acc[0][1], false, false);
      acc[0][2] = __builtin_amdgcn_wmma_f32_16x16x32_f16(false, a0, false, bf2, (short)0, acc[0][2], false, false);
      acc[0][3] = __builtin_amdgcn_wmma_f32_16x16x32_f16(false, a0, false, bf3, (short)0, acc[0][3], false, false);
      acc[1][0] = __builtin_amdgcn_wmma_f32_16x16x32_f16(false, a1, false, bf0, (short)0, acc[1][0], false, false);
      acc[1][1] = __builtin_amdgcn_wmma_f32_16x16x32_f16(false, a1, false, bf1, (short)0, acc[1][1], false, false);
      acc[1][2] = __builtin_amdgcn_wmma_f32_16x16x32_f16(false, a1, false, bf2, (short)0, acc[1][2], false, false);
      acc[1][3] = __builtin_amdgcn_wmma_f32_16x16x32_f16(false, a1, false, bf3, (short)0, acc[1][3], false, false);
    }
  }

  // Epilogue: D element (m,n): spatial = r + 8*h (+16 for tile 1), oc = nt*16 + row
  half_t* outh = (half_t*)outp;
  float*  outf = (float*)outp;
  size_t colv = bbase + (size_t)((x * 32 + y) * 32);
#pragma unroll
  for (int m = 0; m < 2; ++m) {
#pragma unroll
    for (int nt = 0; nt < 4; ++nt) {
      int oc = nt * 16 + row;
      float sc = scale[oc], sh = shift[oc];
#pragma unroll
      for (int r = 0; r < 8; ++r) {
        size_t vg = colv + m * 16 + r + 8 * h;
        float val = acc[m][nt][r] * sc + sh;
        val = val > 0.0f ? val : val * SLOPE;
        if (OUT_F16) outh[vg * 64 + oc] = (half_t)val;
        else         outf[vg * 64 + oc] = val;
      }
    }
  }
}

// ---------------------------------------------------------------------------
// Devoxelize (trilinear gather) + point MLP branch, coalesced [B,C,N] store
// ---------------------------------------------------------------------------
__global__ __launch_bounds__(256) void pv_devox_point(const float* __restrict__ y2,
                                                      const float* __restrict__ featT,
                                                      const float* __restrict__ coords,
                                                      const float* __restrict__ stats,
                                                      const float* __restrict__ mlp_w,
                                                      const float* __restrict__ pscale,
                                                      const float* __restrict__ pshift,
                                                      float* __restrict__ out) {
  __shared__ float sfeat[64 * 64];
  __shared__ float sout[64 * 64];
  int t  = threadIdx.x;
  int g0 = blockIdx.x * 64;                 // 64 points per block
  int b  = g0 / NPTS;
  int n0 = g0 - b * NPTS;
#pragma unroll
  for (int k = 0; k < 16; ++k) {
    int e = k * 256 + t;
    sfeat[e] = featT[(size_t)g0 * 64 + e];
  }
  __syncthreads();

  int p = t & 63, cblk = t >> 6;
  Corners cr; compute_corners(coords, stats, b, n0 + p, cr);
  float res[16];
#pragma unroll
  for (int j = 0; j < 16; ++j) res[j] = 0.0f;
#pragma unroll
  for (int k = 0; k < 8; ++k) {
    const float* gb = y2 + ((size_t)(b * R3 + cr.idx[k])) * 64 + cblk * 16;
    float w = cr.w[k];
#pragma unroll
    for (int j = 0; j < 16; ++j) res[j] += w * gb[j];
  }
  const float* fv = sfeat + p * 64;
  for (int j = 0; j < 16; ++j) {
    int c = cblk * 16 + j;
    const float* wr = mlp_w + (size_t)c * 64;
    float a = 0.0f;
#pragma unroll
    for (int i = 0; i < 64; ++i) a += wr[i] * fv[i];
    a = a * pscale[c] + pshift[c];
    a = fmaxf(a, 0.0f);
    sout[c * 64 + p] = res[j] + a;
  }
  __syncthreads();

  int c = t >> 2, chunk = t & 3;
  size_t obase = (size_t)b * 64 * NPTS + (size_t)c * NPTS + n0 + chunk * 16;
#pragma unroll
  for (int i = 0; i < 16; ++i) out[obase + i] = sout[c * 64 + chunk * 16 + i];
}

// ---------------------------------------------------------------------------
// Host-side orchestration
// ---------------------------------------------------------------------------
extern "C" void kernel_launch(void* const* d_in, const int* in_sizes, int n_in,
                              void* d_out, int out_size, void* d_ws, size_t ws_size,
                              hipStream_t stream) {
  const float* features = (const float*)d_in[0];
  const float* coords   = (const float*)d_in[1];
  const float* conv1_w  = (const float*)d_in[2];
  const float* conv1_b  = (const float*)d_in[3];
  const float* bn1_g = (const float*)d_in[4];
  const float* bn1_b = (const float*)d_in[5];
  const float* bn1_m = (const float*)d_in[6];
  const float* bn1_v = (const float*)d_in[7];
  const float* conv2_w  = (const float*)d_in[8];
  const float* conv2_b  = (const float*)d_in[9];
  const float* bn2_g = (const float*)d_in[10];
  const float* bn2_b = (const float*)d_in[11];
  const float* bn2_m = (const float*)d_in[12];
  const float* bn2_v = (const float*)d_in[13];
  const float* mlp_w = (const float*)d_in[14];
  const float* mlp_b = (const float*)d_in[15];
  const float* bnp_g = (const float*)d_in[16];
  const float* bnp_b = (const float*)d_in[17];
  const float* bnp_m = (const float*)d_in[18];
  const float* bnp_v = (const float*)d_in[19];

  char* ws = (char*)d_ws;
  size_t off = 0;
  auto alloc = [&](size_t bytes) -> char* {
    char* p = ws + off;
    off = (off + bytes + 255) & ~(size_t)255;
    return p;
  };
  float*  featT  = (float*)alloc((size_t)BB * NPTS * 64 * 4);   // 64 MB
  float*  accbuf = (float*)alloc((size_t)BB * R3 * 64 * 4);     // 64 MB (reused as y2)
  float*  cnt    = (float*)alloc((size_t)BB * R3 * 4);          // 1 MB
  half_t* vox    = (half_t*)alloc((size_t)BB * R3 * 64 * 2);    // 32 MB
  half_t* y1     = (half_t*)alloc((size_t)BB * R3 * 64 * 2);    // 32 MB
  half_t* wf1    = (half_t*)alloc((size_t)WFRAG_BYTES);
  half_t* wf2    = (half_t*)alloc((size_t)WFRAG_BYTES);
  float*  stats  = (float*)alloc((size_t)BB * 4 * 4);
  float*  scp    = (float*)alloc((size_t)6 * 64 * 4);

  hipMemsetAsync(accbuf, 0, (size_t)BB * R3 * 64 * 4, stream);
  hipMemsetAsync(cnt,    0, (size_t)BB * R3 * 4,      stream);

  pv_transpose<<<BB * (NPTS / 32), 256, 0, stream>>>(features, featT);
  pv_stats<<<BB, 256, 0, stream>>>(coords, stats);
  pv_bnfold<<<1, 64, 0, stream>>>(conv1_b, bn1_g, bn1_b, bn1_m, bn1_v,
                                  conv2_b, bn2_g, bn2_b, bn2_m, bn2_v,
                                  mlp_b,   bnp_g, bnp_b, bnp_m, bnp_v, scp);
  pv_wswizzle<<<(WFRAG_HALFS + 255) / 256, 256, 0, stream>>>(conv1_w, wf1);
  pv_wswizzle<<<(WFRAG_HALFS + 255) / 256, 256, 0, stream>>>(conv2_w, wf2);

  pv_scatter<<<BB * NPTS / 4, 256, 0, stream>>>(featT, coords, stats, accbuf, cnt);
  pv_normalize<<<(BB * R3 * 64) / 256, 256, 0, stream>>>(accbuf, cnt, vox);

  // 8192 (b,x,y) columns, 8 waves/block -> 1024 blocks, 216KB dynamic LDS
  pv_conv3d<true ><<<BB * 1024 / 8, 256, WFRAG_BYTES, stream>>>(vox, wf1, scp,       scp + 64,  (void*)y1);
  pv_conv3d<false><<<BB * 1024 / 8, 256, WFRAG_BYTES, stream>>>(y1,  wf2, scp + 128, scp + 192, (void*)accbuf);

  pv_devox_point<<<BB * NPTS / 64, 256, 0, stream>>>(accbuf, featT, coords, stats,
                                                     mlp_w, scp + 256, scp + 320,
                                                     (float*)d_out);
  // append coords to output tail (tuple: fused, coords)
  hipMemcpyAsync((float*)d_out + (size_t)BB * 64 * NPTS, coords,
                 (size_t)BB * 3 * NPTS * 4, hipMemcpyDeviceToDevice, stream);
}
